// LatticeSelfAttention_10754598109219
// MI455X (gfx1250) — compile-verified, matrix operand order checked by reading, not run on hardware
//
#include <hip/hip_runtime.h>
#include <hip/hip_bf16.h>

// Problem constants
#define B_  4
#define L_  1024
#define H_  1024
#define NH_ 16
#define HD_ 64
#define E_  8192

typedef __bf16 bf16_t;
typedef __attribute__((ext_vector_type(16))) __bf16 v16bf;
typedef __attribute__((ext_vector_type(8)))  float  v8f;

// K-index for the 16-bit WMMA A/B fragment layout (ISA 7.12.2):
// lanes 0-15: VGPR v<4 -> K=2v,2v+1 ; v>=4 -> K=16+2(v-4)
// lanes 16-31: same +8
__device__ __forceinline__ int frag_k(int v, int hi) {
    int k = (v < 4) ? (2 * v) : (16 + 2 * (v - 4));
    return k + (hi ? 8 : 0);
}

// Load a 16-element (K=32) bf16 fragment whose K-values are contiguous in memory.
// p must already be offset to this lane's row/column start.
__device__ __forceinline__ v16bf load_frag_contig(const bf16_t* p, int hi) {
    v16bf f;
#pragma unroll
    for (int v = 0; v < 8; ++v) {
        int k = frag_k(v, hi);
        f[2 * v]     = p[k];
        f[2 * v + 1] = p[k + 1];
    }
    return f;
}

// Same, but K-values are strided in memory (column of a row-major matrix).
__device__ __forceinline__ v16bf load_frag_strided(const bf16_t* p, int stride, int hi) {
    v16bf f;
#pragma unroll
    for (int v = 0; v < 8; ++v) {
        int k = frag_k(v, hi);
        f[2 * v]     = p[(size_t)k * stride];
        f[2 * v + 1] = p[(size_t)(k + 1) * stride];
    }
    return f;
}

// ---------------------------------------------------------------------------
// bias[b][q][k] = (mask[b][k]==0 ? -1e9 : 0)
__global__ void bias_init_kernel(float* __restrict__ bias, const int* __restrict__ mask) {
    size_t i = (size_t)blockIdx.x * blockDim.x + threadIdx.x;
    size_t total = (size_t)B_ * L_ * L_;
    if (i >= total) return;
    int k = (int)(i % L_);
    int b = (int)(i / ((size_t)L_ * L_));
    bias[i] = (mask[b * L_ + k] == 0) ? -1.0e9f : 0.0f;
}

// scores[b, :, ch, word] += 1.5  (head-independent -> one bias matrix per batch)
__global__ void bias_scatter_kernel(float* __restrict__ bias,
                                    const int* __restrict__ lb,
                                    const int* __restrict__ lch,
                                    const int* __restrict__ lw) {
    int e = blockIdx.x * blockDim.x + threadIdx.x;
    if (e >= E_) return;
    size_t off = ((size_t)lb[e] * L_ + (size_t)lch[e]) * L_ + (size_t)lw[e];
    atomicAdd(bias + off, 1.5f);
}

__global__ void cvt_f32_bf16_kernel(const float* __restrict__ in,
                                    bf16_t* __restrict__ out, size_t n) {
    size_t i = (size_t)blockIdx.x * blockDim.x + threadIdx.x;
    if (i < n) out[i] = (bf16_t)in[i];
}

// ---------------------------------------------------------------------------
// GEMM: Out[m][n] = sum_k A[m][k] * W[n][k] + bias[n]
// A:[M,K] bf16 row-major, W:[N,K] bf16 row-major (i.e. x @ W.T), bias:[N] f32.
// Block = 8 waves = 64(M) x 256(N); each wave computes 32(M) x 64(N).
// B tile (256 cols x 32 K = 16KB) is DMA-staged into LDS with
// GLOBAL_LOAD_ASYNC_TO_LDS_B128, double-buffered on ASYNCcnt.
template <bool OUT_BF16>
__global__ void gemm_wmma_kernel(const bf16_t* __restrict__ A,
                                 const bf16_t* __restrict__ W,
                                 const float* __restrict__ bias,
                                 void* __restrict__ Out,
                                 int M, int N, int K) {
    __shared__ bf16_t bsm[2][256 * 32];   // 2 x 16KB double buffer

    int lane = threadIdx.x & 31;
    int wave = threadIdx.x >> 5;
    int hi   = lane >> 4;
    int nsub = lane & 15;

    int m0     = blockIdx.x * 64 + (wave >> 2) * 32;      // wave M base (2 subtiles)
    int n0base = blockIdx.y * 256;                         // block N base
    int colb   = (wave & 3) * 64;                          // wave N base within block

    // LDS byte address of the staging buffer (flat shared addr low 32 bits = LDS offset)
    unsigned lds_base = (unsigned)(uintptr_t)(&bsm[0][0]);

    // each thread stages 4 x 16B: idx in [0,1024), col = idx>>2, sub = idx&3
    // LDS layout: [col][k] rows of 32 bf16 (64B) per column
    unsigned sidx = threadIdx.x;

    // issue async stage of one k-slice into buffer `buf`
    auto stage = [&](int buf, int kk) {
#pragma unroll
        for (int r = 0; r < 4; ++r) {
            unsigned idx = sidx + r * 256u;
            unsigned col = idx >> 2;
            unsigned sub = idx & 3u;
            unsigned goff = (unsigned)(((n0base + col) * (unsigned)K + (unsigned)kk) * 2u + sub * 16u);
            unsigned laddr = lds_base + (unsigned)buf * (256u * 32u * 2u) + col * 64u + sub * 16u;
            asm volatile("global_load_async_to_lds_b128 %0, %1, %2"
                         :
                         : "v"(laddr), "v"(goff), "s"(W)
                         : "memory");
        }
    };

    v8f c[2][4] = {};
    const bf16_t* arow0 = A + (size_t)(m0 + nsub) * K;
    const bf16_t* arow1 = A + (size_t)(m0 + 16 + nsub) * K;

    stage(0, 0);
    int ib = 0;
    for (int kk = 0; kk < K; kk += 32, ib ^= 1) {
        if (kk + 32 < K) {
            stage(ib ^ 1, kk + 32);
            // in-order ASYNCcnt: <=4 outstanding means current buffer has landed
            asm volatile("s_wait_asynccnt 0x4" ::: "memory");
        } else {
            asm volatile("s_wait_asynccnt 0x0" ::: "memory");
        }
        __syncthreads();

        __builtin_prefetch(arow0 + kk + 128, 0, 3);
        __builtin_prefetch(arow1 + kk + 128, 0, 3);
        v16bf a0 = load_frag_contig(arow0 + kk, hi);
        v16bf a1 = load_frag_contig(arow1 + kk, hi);
#pragma unroll
        for (int j = 0; j < 4; ++j) {
            const bf16_t* bcol = &bsm[ib][(colb + j * 16 + nsub) * 32];
            v16bf bf = load_frag_contig(bcol, hi);
            c[0][j] = __builtin_amdgcn_wmma_f32_16x16x32_bf16(
                false, a0, false, bf, (short)0, c[0][j], false, false);
            c[1][j] = __builtin_amdgcn_wmma_f32_16x16x32_bf16(
                false, a1, false, bf, (short)0, c[1][j], false, false);
        }
        __syncthreads();   // protect buffer ib from next-next stage
    }

    int mo = hi * 8;
#pragma unroll
    for (int mt = 0; mt < 2; ++mt) {
        int mbase = m0 + mt * 16 + mo;
#pragma unroll
        for (int j = 0; j < 4; ++j) {
            int n = n0base + colb + j * 16 + nsub;
            float bb = bias[n];
#pragma unroll
            for (int r = 0; r < 8; ++r) {
                float val = c[mt][j][r] + bb;
                size_t off = (size_t)(mbase + r) * N + n;
                if constexpr (OUT_BF16)
                    ((bf16_t*)Out)[off] = (bf16_t)val;
                else
                    ((float*)Out)[off] = val;
            }
        }
    }
}

// ---------------------------------------------------------------------------
// Attention: block = 4 waves, each wave handles one 16-row q-tile of one (b,h).
// LDS: 16x1024 f32 score strip per wave (64KB x 4 = 256KB of the 320KB WGP LDS).
__global__ void attention_kernel(const bf16_t* __restrict__ Q,
                                 const bf16_t* __restrict__ Km,
                                 const bf16_t* __restrict__ V,
                                 const float* __restrict__ bias,
                                 bf16_t* __restrict__ Ctx) {
    extern __shared__ float smem[];

    int lane = threadIdx.x & 31;
    int wave = threadIdx.x >> 5;
    int hi   = lane >> 4;
    int nsub = lane & 15;
    int mo   = hi * 8;

    int bh   = blockIdx.x >> 4;    // (b*NH + h)
    int qblk = blockIdx.x & 15;
    int b    = bh >> 4;
    int h    = bh & 15;
    int q0   = qblk * 64 + wave * 16;

    float* srow = smem + (size_t)wave * 16 * L_;
    const float scale = 0.125f;    // 1/sqrt(HD)

    // Q fragments for this 16-row tile: K-dim = HD = 64 -> two k-steps of 32
    const bf16_t* qptr = Q + ((size_t)(b * L_ + q0 + nsub)) * H_ + h * HD_;
    v16bf a0 = load_frag_contig(qptr, hi);
    v16bf a1 = load_frag_contig(qptr + 32, hi);

    // ---- scores = scale * Q K^T + bias  -> LDS
    for (int kt = 0; kt < L_ / 16; ++kt) {
        int k0 = kt * 16;
        const bf16_t* kptr = Km + ((size_t)(b * L_ + k0 + nsub)) * H_ + h * HD_;
        v16bf b0 = load_frag_contig(kptr, hi);
        v16bf b1 = load_frag_contig(kptr + 32, hi);
        v8f cs = {};
        cs = __builtin_amdgcn_wmma_f32_16x16x32_bf16(false, a0, false, b0, (short)0, cs, false, false);
        cs = __builtin_amdgcn_wmma_f32_16x16x32_bf16(false, a1, false, b1, (short)0, cs, false, false);
#pragma unroll
        for (int r = 0; r < 8; ++r) {
            int q = q0 + mo + r;
            float sc = cs[r] * scale + bias[((size_t)(b * L_ + q)) * L_ + k0 + nsub];
            srow[(mo + r) * L_ + k0 + nsub] = sc;
        }
    }
    __syncthreads();

    // ---- row-wise softmax (in place, f32), wave32 shuffle reductions
    for (int r = 0; r < 16; ++r) {
        float* row = srow + r * L_;
        float mx = -3.0e38f;
        for (int cc = lane; cc < L_; cc += 32) mx = fmaxf(mx, row[cc]);
#pragma unroll
        for (int o = 16; o > 0; o >>= 1) mx = fmaxf(mx, __shfl_xor(mx, o, 32));
        float sum = 0.0f;
        for (int cc = lane; cc < L_; cc += 32) {
            float e = __expf(row[cc] - mx);
            row[cc] = e;
            sum += e;
        }
#pragma unroll
        for (int o = 16; o > 0; o >>= 1) sum += __shfl_xor(sum, o, 32);
        float inv = 1.0f / sum;
        for (int cc = lane; cc < L_; cc += 32) row[cc] *= inv;
    }
    __syncthreads();

    // ---- context = P @ V   (4 output d-tiles of 16, K-loop over 1024 in 32s)
#pragma unroll 1
    for (int j = 0; j < 4; ++j) {
        v8f c = {};
        for (int kk = 0; kk < L_; kk += 32) {
            // A fragment from probs in LDS (f32 -> bf16)
            const float* prow = srow + nsub * L_ + kk;
            v16bf pa;
#pragma unroll
            for (int v = 0; v < 8; ++v) {
                int k = frag_k(v, hi);
                pa[2 * v]     = (bf16_t)prow[k];
                pa[2 * v + 1] = (bf16_t)prow[k + 1];
            }
            // B fragment: column d of V (stride H)
            const bf16_t* vptr = V + ((size_t)(b * L_ + kk)) * H_ + h * HD_ + j * 16 + nsub;
            v16bf pb = load_frag_strided(vptr, H_, hi);
            c = __builtin_amdgcn_wmma_f32_16x16x32_bf16(
                false, pa, false, pb, (short)0, c, false, false);
        }
#pragma unroll
        for (int r = 0; r < 8; ++r) {
            Ctx[((size_t)(b * L_ + q0 + mo + r)) * H_ + h * HD_ + j * 16 + nsub] = (bf16_t)c[r];
        }
    }
}

// ---------------------------------------------------------------------------
extern "C" void kernel_launch(void* const* d_in, const int* in_sizes, int n_in,
                              void* d_out, int out_size, void* d_ws, size_t ws_size,
                              hipStream_t stream) {
    const float* hs  = (const float*)d_in[0];
    const float* Wq  = (const float*)d_in[1];
    const float* bq  = (const float*)d_in[2];
    const float* Wk  = (const float*)d_in[3];
    const float* bk  = (const float*)d_in[4];
    const float* Wv  = (const float*)d_in[5];
    const float* bv  = (const float*)d_in[6];
    const float* Wo  = (const float*)d_in[7];
    const float* bo  = (const float*)d_in[8];
    const int* mask  = (const int*)d_in[9];
    const int* lb    = (const int*)d_in[10];
    const int* lch   = (const int*)d_in[11];
    const int* lw    = (const int*)d_in[12];

    char* ws = (char*)d_ws;
    size_t off = 0;
    float*  bias = (float*)(ws + off);  off += (size_t)B_ * L_ * L_ * sizeof(float);   // 16 MB
    bf16_t* Xbf  = (bf16_t*)(ws + off); off += (size_t)B_ * L_ * H_ * 2;               //  8 MB
    bf16_t* Wqb  = (bf16_t*)(ws + off); off += (size_t)H_ * H_ * 2;
    bf16_t* Wkb  = (bf16_t*)(ws + off); off += (size_t)H_ * H_ * 2;
    bf16_t* Wvb  = (bf16_t*)(ws + off); off += (size_t)H_ * H_ * 2;
    bf16_t* Wob  = (bf16_t*)(ws + off); off += (size_t)H_ * H_ * 2;                    //  8 MB total
    bf16_t* Qb   = (bf16_t*)(ws + off); off += (size_t)B_ * L_ * H_ * 2;
    bf16_t* Kb   = (bf16_t*)(ws + off); off += (size_t)B_ * L_ * H_ * 2;
    bf16_t* Vb   = (bf16_t*)(ws + off); off += (size_t)B_ * L_ * H_ * 2;               // 24 MB
    bf16_t* Cb   = (bf16_t*)(ws + off); off += (size_t)B_ * L_ * H_ * 2;               //  8 MB

    // 1) score bias matrix (mask + lexicon scatter)
    {
        size_t total = (size_t)B_ * L_ * L_;
        bias_init_kernel<<<(unsigned)((total + 255) / 256), 256, 0, stream>>>(bias, mask);
        bias_scatter_kernel<<<(E_ + 255) / 256, 256, 0, stream>>>(bias, lb, lch, lw);
    }

    // 2) fp32 -> bf16 conversions
    {
        size_t nx = (size_t)B_ * L_ * H_;
        cvt_f32_bf16_kernel<<<(unsigned)((nx + 255) / 256), 256, 0, stream>>>(hs, Xbf, nx);
        size_t nw = (size_t)H_ * H_;
        unsigned gw = (unsigned)((nw + 255) / 256);
        cvt_f32_bf16_kernel<<<gw, 256, 0, stream>>>(Wq, Wqb, nw);
        cvt_f32_bf16_kernel<<<gw, 256, 0, stream>>>(Wk, Wkb, nw);
        cvt_f32_bf16_kernel<<<gw, 256, 0, stream>>>(Wv, Wvb, nw);
        cvt_f32_bf16_kernel<<<gw, 256, 0, stream>>>(Wo, Wob, nw);
    }

    // 3) QKV projections (M=4096, N=1024, K=1024)
    {
        dim3 g(4096 / 64, 1024 / 256);
        gemm_wmma_kernel<true><<<g, 256, 0, stream>>>(Xbf, Wqb, bq, Qb, 4096, 1024, 1024);
        gemm_wmma_kernel<true><<<g, 256, 0, stream>>>(Xbf, Wkb, bk, Kb, 4096, 1024, 1024);
        gemm_wmma_kernel<true><<<g, 256, 0, stream>>>(Xbf, Wvb, bv, Vb, 4096, 1024, 1024);
    }

    // 4) attention (B*NH*(L/64) blocks, 4 waves each, 256KB dynamic LDS)
    {
        unsigned grid = B_ * NH_ * (L_ / 64);             // 1024
        size_t lds = (size_t)4 * 16 * L_ * sizeof(float); // 262144
        attention_kernel<<<grid, 128, (unsigned)lds, stream>>>(Qb, Kb, Vb, bias, Cb);
    }

    // 5) output projection -> d_out (f32)
    {
        dim3 g(4096 / 64, 1024 / 256);
        gemm_wmma_kernel<false><<<g, 256, 0, stream>>>(Cb, Wob, bo, d_out, 4096, 1024, 1024);
    }
}
